// PartialFC_SingleGPU_57509612093618
// MI455X (gfx1250) — matrix operand order repaired and checked by reading, not compiled, not run on hardware
//
#include <hip/hip_runtime.h>
#include <hip/hip_bf16.h>
#include <math.h>

typedef __attribute__((ext_vector_type(16))) _Float16 v16h;
typedef __attribute__((ext_vector_type(8)))  _Float16 v8h;
typedef __attribute__((ext_vector_type(8)))  float    v8f;

#define EMBED   512
#define NCLS    100000
#define BATCH   512
#define ARC_S   64.0f
#define ARC_M   0.5f
#define ARC_EPS 1e-7f

// ---------------------------------------------------------------------------
// Kernel 1: L2-normalize embeddings rows, store as f16 (row-major [512][512]).
// One block per row, 256 threads, 2 elements per thread.
// ---------------------------------------------------------------------------
__global__ __launch_bounds__(256) void k_norm_emb(const float* __restrict__ E,
                                                  _Float16* __restrict__ A)
{
    const int b   = blockIdx.x;
    const int tid = threadIdx.x;
    __shared__ float red[256];

    const float v0 = E[(size_t)b * EMBED + tid];
    const float v1 = E[(size_t)b * EMBED + tid + 256];
    red[tid] = v0 * v0 + v1 * v1;
    __syncthreads();
    for (int off = 128; off > 0; off >>= 1) {
        if (tid < off) red[tid] += red[tid + off];
        __syncthreads();
    }
    const float inv = 1.0f / fmaxf(sqrtf(red[0]), 1e-12f);
    A[(size_t)b * EMBED + tid]       = (_Float16)(v0 * inv);
    A[(size_t)b * EMBED + tid + 256] = (_Float16)(v1 * inv);
}

// ---------------------------------------------------------------------------
// Kernel 2: WMMA GEMM + ArcFace epilogue.
//   logits[b, c] = s * margin(clip(dot(e_hat_b, w_c) / ||w_c||, -1, 1))
// Grid: 6250 blocks (one per 16-class tile), 256 threads = 8 waves.
// Each wave: 4 M-tiles (rows wave*64 .. wave*64+63) x 1 N-tile, K = 512.
// Weight rows are loaded f32 (coalesced float4), converted to f16 in-register;
// sum of squares accumulated on the fly for the 1/||w|| epilogue scale.
// ---------------------------------------------------------------------------
__global__ __launch_bounds__(256) void k_gemm_arcface(const _Float16* __restrict__ A,
                                                      const float* __restrict__ W,
                                                      const int* __restrict__ labels,
                                                      float* __restrict__ logits)
{
    const int lane  = threadIdx.x & 31;
    const int wave  = threadIdx.x >> 5;
    const int khalf = lane >> 4;      // 0: K low half, 1: K high half
    const int mlane = lane & 15;      // row within M-tile (A), col within N-tile (B/D)
    const int n0    = blockIdx.x * 16;
    const int nlane = n0 + mlane;     // this lane's class id

    const float* wrow = W + (size_t)nlane * EMBED + khalf * 16;
    const int mbase = wave * 64;
    const int koffA = khalf * 16;

    v8f acc0 = {}, acc1 = {}, acc2 = {}, acc3 = {};
    float wss = 0.0f;

    for (int k0 = 0; k0 < EMBED; k0 += 32) {
        // ---- B fragment: 16 contiguous f32 of class row 'nlane', K = k0 + khalf*16 ...
        v16h bfrag;
        const float4* wp = (const float4*)(wrow + k0);
        if (k0 + 32 < EMBED) __builtin_prefetch(wrow + k0 + 32, 0, 0);
        #pragma unroll
        for (int q = 0; q < 4; ++q) {
            const float4 f = wp[q];
            wss += f.x * f.x + f.y * f.y + f.z * f.z + f.w * f.w;
            bfrag[q * 4 + 0] = (_Float16)f.x;
            bfrag[q * 4 + 1] = (_Float16)f.y;
            bfrag[q * 4 + 2] = (_Float16)f.z;
            bfrag[q * 4 + 3] = (_Float16)f.w;
        }
        // ---- 4 A fragments + WMMAs
        #pragma unroll
        for (int t = 0; t < 4; ++t) {
            const int m = mbase + t * 16 + mlane;
            const _Float16* ap = A + (size_t)m * EMBED + k0 + koffA;
            const v8h lo = *(const v8h*)ap;
            const v8h hi = *(const v8h*)(ap + 8);
            const v16h afrag = __builtin_shufflevector(lo, hi,
                0, 1, 2, 3, 4, 5, 6, 7, 8, 9, 10, 11, 12, 13, 14, 15);
            v8f c = (t == 0) ? acc0 : (t == 1) ? acc1 : (t == 2) ? acc2 : acc3;
            c = __builtin_amdgcn_wmma_f32_16x16x32_f16(
                    /*neg_a=*/false, afrag, /*neg_b=*/false, bfrag,
                    /*c_mod=*/(short)0, c, /*reuse_a=*/false, /*reuse_b=*/false);
            if (t == 0) acc0 = c; else if (t == 1) acc1 = c; else if (t == 2) acc2 = c; else acc3 = c;
        }
    }

    // lanes l and l^16 each hold half of class nlane's ||w||^2
    wss += __shfl_xor(wss, 16, 32);
    const float invw = 1.0f / fmaxf(sqrtf(wss), 1e-12f);

    // ---- epilogue: D layout -> m_local = r + 8*khalf, n = mlane
    #pragma unroll
    for (int t = 0; t < 4; ++t) {
        const v8f c = (t == 0) ? acc0 : (t == 1) ? acc1 : (t == 2) ? acc2 : acc3;
        #pragma unroll
        for (int r = 0; r < 8; ++r) {
            const int m = mbase + t * 16 + r + khalf * 8;
            float cosv = c[r] * invw;
            cosv = fminf(fmaxf(cosv, -1.0f), 1.0f);
            if (labels[m] == nlane) {
                const float tc = fminf(fmaxf(cosv, -1.0f + ARC_EPS), 1.0f - ARC_EPS);
                cosv = cosf(acosf(tc) + ARC_M);
            }
            logits[(size_t)m * NCLS + nlane] = cosv * ARC_S;
        }
    }
}

// ---------------------------------------------------------------------------
// Kernel 3: per-row online softmax stats + target log-prob. One block per row.
// ---------------------------------------------------------------------------
__global__ __launch_bounds__(256) void k_row_logp(const float* __restrict__ logits,
                                                  const int* __restrict__ labels,
                                                  float* __restrict__ row_logp)
{
    const int b   = blockIdx.x;
    const int tid = threadIdx.x;
    const float* row = logits + (size_t)b * NCLS;

    float m = -INFINITY, s = 0.0f;
    for (int i = tid; i < NCLS; i += 256) {
        const float x = row[i];
        if (x > m) { s = s * __expf(m - x) + 1.0f; m = x; }
        else       { s += __expf(x - m); }
    }

    __shared__ float rm[256], rs[256];
    rm[tid] = m; rs[tid] = s;
    __syncthreads();
    for (int off = 128; off > 0; off >>= 1) {
        if (tid < off) {
            const float m2 = rm[tid + off], s2 = rs[tid + off];
            const float M = fmaxf(rm[tid], m2);
            rs[tid] = rs[tid] * __expf(rm[tid] - M) + s2 * __expf(m2 - M);
            rm[tid] = M;
        }
        __syncthreads();
    }
    if (tid == 0) {
        const int lbl = labels[b];
        const float t = row[lbl];
        row_logp[b] = t - rm[0] - logf(rs[0]);
    }
}

// ---------------------------------------------------------------------------
// Kernel 4: deterministic reduction of 512 per-row log-probs -> loss.
// ---------------------------------------------------------------------------
__global__ __launch_bounds__(256) void k_loss(const float* __restrict__ row_logp,
                                              float* __restrict__ out)
{
    const int tid = threadIdx.x;
    __shared__ float red[256];
    red[tid] = row_logp[tid] + row_logp[tid + 256];
    __syncthreads();
    for (int off = 128; off > 0; off >>= 1) {
        if (tid < off) red[tid] += red[tid + off];
        __syncthreads();
    }
    if (tid == 0) out[0] = -red[0] / (float)BATCH;
}

// ---------------------------------------------------------------------------
extern "C" void kernel_launch(void* const* d_in, const int* in_sizes, int n_in,
                              void* d_out, int out_size, void* d_ws, size_t ws_size,
                              hipStream_t stream)
{
    const float* emb    = (const float*)d_in[0];   // [512, 512] f32
    const int*   labels = (const int*)  d_in[1];   // [512] int
    const float* weight = (const float*)d_in[2];   // [100000, 512] f32

    float* out    = (float*)d_out;                 // [0] = loss, [1..] = logits
    float* logits = out + 1;

    _Float16* A_f16    = (_Float16*)d_ws;                                  // 512 KB
    float*    row_logp = (float*)((char*)d_ws + (size_t)BATCH * EMBED * 2); // 2 KB

    k_norm_emb   <<<BATCH,     256, 0, stream>>>(emb, A_f16);
    k_gemm_arcface<<<NCLS / 16, 256, 0, stream>>>(A_f16, weight, labels, logits);
    k_row_logp   <<<BATCH,     256, 0, stream>>>(logits, labels, row_logp);
    k_loss       <<<1,         256, 0, stream>>>(row_logp, out);
}
